// GraphSage_60490319397131
// MI455X (gfx1250) — matrix operand order repaired and checked by reading, not compiled, not run on hardware
//
#include <hip/hip_runtime.h>

// ---------------------------------------------------------------------------
// CDNA5 GraphSAGE forward. Strategy: bf16 WMMA (v_wmma_f32_16x16x32_bf16)
// for all GEMMs, fp32 accumulate. dif_mat (fp32) converted to bf16 in-register
// while streaming from HBM; B operands pre-transposed to bf16 in workspace.
// LDS double-buffered B tiles shared across the 8 waves of a workgroup.
// This round: B fragments preloaded into distinct registers (batched ds_loads,
// pipelined waits) and A stream software-pipelined one k-step ahead.
// ---------------------------------------------------------------------------

typedef __attribute__((ext_vector_type(16))) __bf16        v16bf;
typedef __attribute__((ext_vector_type(8)))  float         v8f;
typedef __attribute__((ext_vector_type(8)))  unsigned int  v8u;

// round-to-nearest-even fp32 -> bf16 (bit form)
__device__ __forceinline__ unsigned short f2bf(float x) {
    unsigned u = __float_as_uint(x);
    u += 0x7FFFu + ((u >> 16) & 1u);
    return (unsigned short)(u >> 16);
}
// pack two fp32 into one dword of bf16 pair (lo = a, hi = b)
__device__ __forceinline__ unsigned int pkbf(float a, float b) {
    unsigned ua = __float_as_uint(a);
    unsigned ub = __float_as_uint(b);
    ua += 0x7FFFu + ((ua >> 16) & 1u);
    ub += 0x7FFFu + ((ub >> 16) & 1u);
    return (ua >> 16) | (ub & 0xFFFF0000u);
}

// Convert 4 float4 (A raw, K = {0..7} and {16..23} relative) to an A fragment.
__device__ __forceinline__ v16bf cvtA(float4 f0, float4 f1, float4 f2, float4 f3) {
    v8u r;
    r[0] = pkbf(f0.x, f0.y); r[1] = pkbf(f0.z, f0.w);
    r[2] = pkbf(f1.x, f1.y); r[3] = pkbf(f1.z, f1.w);
    r[4] = pkbf(f2.x, f2.y); r[5] = pkbf(f2.z, f2.w);
    r[6] = pkbf(f3.x, f3.y); r[7] = pkbf(f3.z, f3.w);
    return __builtin_bit_cast(v16bf, r);
}

// A-fragment from bf16 row data. p = row base + k + half*8.
// lanes 0-15: VGPR0-3 = K0..7, VGPR4-7 = K16..23 ; lanes 16-31: +8.
__device__ __forceinline__ v16bf fragA_bf16(const unsigned short* p) {
    uint4 lo = *(const uint4*)(p);
    uint4 hi = *(const uint4*)(p + 16);
    v8u r;
    r[0] = lo.x; r[1] = lo.y; r[2] = lo.z; r[3] = lo.w;
    r[4] = hi.x; r[5] = hi.y; r[6] = hi.z; r[7] = hi.w;
    return __builtin_bit_cast(v16bf, r);
}

// B-fragment (32x16 bf16) from k-contiguous (transposed) bf16 data.
// lanes 0-15 hold K0..15, lanes 16-31 hold K16..31: p = col base + k + half*16.
__device__ __forceinline__ v16bf fragB_bf16(const unsigned short* p) {
    uint4 lo = *(const uint4*)(p);
    uint4 hi = *(const uint4*)(p + 8);
    v8u r;
    r[0] = lo.x; r[1] = lo.y; r[2] = lo.z; r[3] = lo.w;
    r[4] = hi.x; r[5] = hi.y; r[6] = hi.z; r[7] = hi.w;
    return __builtin_bit_cast(v16bf, r);
}

// ---------------------------------------------------------------------------
// Gather kernels: build transposed bf16 B matrix and fp32 dst (self) features
// ---------------------------------------------------------------------------
// Bt[n][k] = bf16( X[ map2? map2[map1[k]] : map1[k] ][n] ),  X rows of 128
__global__ __launch_bounds__(256) void k_gather_bt(
        const float* __restrict__ X, const int* __restrict__ map1,
        const int* __restrict__ map2, unsigned short* __restrict__ Bt, int K) {
    int k = blockIdx.x * 256 + threadIdx.x;
    if (k >= K) return;
    int row = map1[k];
    if (map2) row = map2[row];
    const float* src = X + (size_t)row * 128;
    for (int n = 0; n < 128; ++n)
        Bt[(size_t)n * K + k] = f2bf(src[n]);     // coalesced across lanes in k
}

// dstbuf[i][n] = X[ map2? map2[map1[i]] : map1[i] ][n]
__global__ __launch_bounds__(256) void k_gather_dst(
        const float* __restrict__ X, const int* __restrict__ map1,
        const int* __restrict__ map2, float* __restrict__ dstbuf, int ndst) {
    int idx = blockIdx.x * 256 + threadIdx.x;
    if (idx >= ndst * 128) return;
    int i = idx >> 7, n = idx & 127;
    int row = map1[i];
    if (map2) row = map2[row];
    dstbuf[idx] = X[(size_t)row * 128 + n];
}

// Wt[n][k] = bf16(W[k][n]);  W is [256][128]
__global__ __launch_bounds__(256) void k_transpose_w(
        const float* __restrict__ W, unsigned short* __restrict__ Wt) {
    int idx = blockIdx.x * 256 + threadIdx.x;
    if (idx >= 256 * 128) return;
    int k = idx >> 7, n = idx & 127;
    Wt[(size_t)n * 256 + k] = f2bf(W[idx]);
}

// ---------------------------------------------------------------------------
// Main WMMA GEMM:  partial[kc] += A(fp32, MxKtot) x Bt(bf16, 128xKtot)^T
// grid = (M/128, num_kchunks), block = 256 (8 waves, 16 rows each)
// ---------------------------------------------------------------------------
__global__ __launch_bounds__(256) void k_gemm_bf16(
        const float* __restrict__ A, const unsigned short* __restrict__ Bt,
        float* __restrict__ part, int M, int Ktot, int kchunk) {
    // padded LDS: 40-ushort rows -> 20-bank stride, conflict-free b128 reads
    __shared__ unsigned short BtL[2][128][40];

    const int t    = threadIdx.x;
    const int wave = t >> 5;
    const int lane = t & 31;
    const int hf   = lane >> 4;
    const int r    = lane & 15;
    const int mrow0 = blockIdx.x * 128 + wave * 16;
    const int kc    = blockIdx.y;
    const int k0    = kc * kchunk;
    const int nsteps = kchunk >> 5;

    v8f acc[8];
#pragma unroll
    for (int i = 0; i < 8; ++i) {
        v8f z = {0.f, 0.f, 0.f, 0.f, 0.f, 0.f, 0.f, 0.f};
        acc[i] = z;
    }

    const float* arow = A + (size_t)(mrow0 + r) * Ktot + hf * 8;

    auto stage = [&](int s, int kb) {
        for (int c = t; c < 512; c += 256) {          // 128 rows x 4 x b128
            int n = c >> 2, q = c & 3;
            *(uint4*)&BtL[s][n][q * 8] =
                *(const uint4*)(Bt + (size_t)n * Ktot + kb + q * 8);
        }
    };

    stage(0, k0);

    // software pipeline: raw A for step 0 in flight while LDS fills
    float4 a0 = *(const float4*)(arow + k0);
    float4 a1 = *(const float4*)(arow + k0 + 4);
    float4 a2 = *(const float4*)(arow + k0 + 16);
    float4 a3 = *(const float4*)(arow + k0 + 20);

    __syncthreads();

    for (int s = 0; s < nsteps; ++s) {
        const int kb  = k0 + (s << 5);
        const int cur = s & 1;

        // stage next B tile into the other LDS buffer
        if (s + 1 < nsteps) stage(cur ^ 1, kb + 32);

        // issue next-step A loads early (hide HBM latency under compute)
        float4 n0, n1, n2, n3;
        if (s + 1 < nsteps) {
            const float* ap = arow + kb + 32;
            n0 = *(const float4*)(ap);
            n1 = *(const float4*)(ap + 4);
            n2 = *(const float4*)(ap + 16);
            n3 = *(const float4*)(ap + 20);
            __builtin_prefetch(ap + 128, 0, 1);       // global_prefetch ahead
        }

        v16bf a = cvtA(a0, a1, a2, a3);               // fp32 -> bf16 convert

        // preload all 8 B fragments (batched ds_loads, pipelined waits)
        v16bf bfr[8];
#pragma unroll
        for (int nt = 0; nt < 8; ++nt) {
            const int nn = nt * 16 + r;
            uint4 lo = *(const uint4*)&BtL[cur][nn][hf * 8];
            uint4 hi = *(const uint4*)&BtL[cur][nn][hf * 8 + 16];
            v8u rb;
            rb[0] = lo.x; rb[1] = lo.y; rb[2] = lo.z; rb[3] = lo.w;
            rb[4] = hi.x; rb[5] = hi.y; rb[6] = hi.z; rb[7] = hi.w;
            bfr[nt] = __builtin_bit_cast(v16bf, rb);
        }

#pragma unroll
        for (int nt = 0; nt < 8; ++nt)
            acc[nt] = __builtin_amdgcn_wmma_f32_16x16x32_bf16(
                false, a, false, bfr[nt], (short)0, acc[nt], false, false);

        a0 = n0; a1 = n1; a2 = n2; a3 = n3;
        __syncthreads();
    }

    float* out = part + ((size_t)kc * M + mrow0) * 128;
#pragma unroll
    for (int nt = 0; nt < 8; ++nt)
#pragma unroll
        for (int g = 0; g < 8; ++g)
            out[(hf * 8 + g) * 128 + nt * 16 + r] = acc[nt][g];
}

// ---------------------------------------------------------------------------
// Fixed-order split-K reduction + concat [agg | dst] -> bf16 A for the MLP
// ---------------------------------------------------------------------------
__global__ __launch_bounds__(256) void k_reduce_cat(
        const float* __restrict__ part, int nkc, const float* __restrict__ dst,
        unsigned short* __restrict__ Acat, int M) {
    int idx = blockIdx.x * 256 + threadIdx.x;
    if (idx >= M * 128) return;
    int mrow = idx >> 7, col = idx & 127;
    float s = 0.f;
    for (int c = 0; c < nkc; ++c)
        s += part[((size_t)c * M + mrow) * 128 + col];
    Acat[(size_t)mrow * 256 + col]       = f2bf(s);
    Acat[(size_t)mrow * 256 + 128 + col] = f2bf(dst[idx]);
}

// ---------------------------------------------------------------------------
// MLP: H = relu(Acat(bf16, Mx256) @ W(bf16 transposed 128x256)), WMMA
// grid = M/128, block = 256
// ---------------------------------------------------------------------------
__global__ __launch_bounds__(256) void k_mlp(
        const unsigned short* __restrict__ Acat,
        const unsigned short* __restrict__ Wt, float* __restrict__ H, int M) {
    const int t    = threadIdx.x;
    const int wave = t >> 5;
    const int lane = t & 31;
    const int hf   = lane >> 4;
    const int r    = lane & 15;
    const int mrow0 = blockIdx.x * 128 + wave * 16;

    v8f acc[8];
#pragma unroll
    for (int i = 0; i < 8; ++i) {
        v8f z = {0.f, 0.f, 0.f, 0.f, 0.f, 0.f, 0.f, 0.f};
        acc[i] = z;
    }

    const unsigned short* arow = Acat + (size_t)(mrow0 + r) * 256 + hf * 8;
#pragma unroll
    for (int k = 0; k < 256; k += 32) {
        v16bf a = fragA_bf16(arow + k);
        v16bf bfr[8];
#pragma unroll
        for (int nt = 0; nt < 8; ++nt)
            bfr[nt] = fragA_bf16(Wt + (size_t)(nt * 16 + r) * 256 + k + hf * 8);
#pragma unroll
        for (int nt = 0; nt < 8; ++nt)
            acc[nt] = __builtin_amdgcn_wmma_f32_16x16x32_bf16(
                false, a, false, bfr[nt], (short)0, acc[nt], false, false);
    }
#pragma unroll
    for (int nt = 0; nt < 8; ++nt)
#pragma unroll
        for (int g = 0; g < 8; ++g)
            H[(size_t)(mrow0 + hf * 8 + g) * 128 + nt * 16 + r] =
                fmaxf(acc[nt][g], 0.f);
}

// ---------------------------------------------------------------------------
// Classifier: out = softmax(H2 @ w_cls), N=2
// ---------------------------------------------------------------------------
__global__ __launch_bounds__(256) void k_cls(
        const float* __restrict__ H2, const float* __restrict__ Wcls,
        float* __restrict__ out, int M) {
    int i = blockIdx.x * 256 + threadIdx.x;
    if (i >= M) return;
    float s0 = 0.f, s1 = 0.f;
    const float* h = H2 + (size_t)i * 128;
    for (int k = 0; k < 128; ++k) {
        float v = h[k];
        s0 += v * Wcls[k * 2 + 0];
        s1 += v * Wcls[k * 2 + 1];
    }
    float mx = fmaxf(s0, s1);
    float e0 = __expf(s0 - mx), e1 = __expf(s1 - mx);
    float inv = 1.f / (e0 + e1);
    out[i * 2 + 0] = e0 * inv;
    out[i * 2 + 1] = e1 * inv;
}

// ---------------------------------------------------------------------------
extern "C" void kernel_launch(void* const* d_in, const int* in_sizes, int n_in,
                              void* d_out, int out_size, void* d_ws, size_t ws_size,
                              hipStream_t stream) {
    (void)in_sizes; (void)n_in; (void)out_size; (void)ws_size;

    const int*   src_nodes = (const int*)  d_in[0];
    const int*   d2s1      = (const int*)  d_in[1];
    const int*   d2d1      = (const int*)  d_in[2];
    const float* dif1      = (const float*)d_in[3];
    const int*   d2s2      = (const int*)  d_in[4];
    const int*   d2d2      = (const int*)  d_in[5];
    const float* dif2      = (const float*)d_in[6];
    const float* raw       = (const float*)d_in[7];
    const float* w1        = (const float*)d_in[8];
    const float* w2        = (const float*)d_in[9];
    const float* wcls      = (const float*)d_in[10];

    char*  ws  = (char*)d_ws;
    size_t off = 0;
    auto alloc = [&](size_t bytes) {
        void* p = ws + off;
        off = (off + bytes + 255) & ~(size_t)255;
        return p;
    };
    unsigned short* Bt1   = (unsigned short*)alloc(128ull * 40000 * 2);
    float*          dst1  = (float*)         alloc(4096ull * 128 * 4);
    float*          part1 = (float*)         alloc(10ull * 4096 * 128 * 4);
    unsigned short* Acat1 = (unsigned short*)alloc(4096ull * 256 * 2);
    unsigned short* w1t   = (unsigned short*)alloc(128ull * 256 * 2);
    unsigned short* w2t   = (unsigned short*)alloc(128ull * 256 * 2);
    float*          h1    = (float*)         alloc(4096ull * 128 * 4);
    unsigned short* Bt2   = (unsigned short*)alloc(128ull * 4096 * 2);
    float*          dst2  = (float*)         alloc(1024ull * 128 * 4);
    float*          part2 = (float*)         alloc(4ull * 1024 * 128 * 4);
    unsigned short* Acat2 = (unsigned short*)alloc(1024ull * 256 * 2);
    float*          h2    = (float*)         alloc(1024ull * 128 * 4);

    // layer 1: gather + big WMMA GEMM (M=4096, K=40000, split-K=10) + MLP
    k_gather_bt <<<(40000 + 255) / 256, 256, 0, stream>>>(raw, d2s1, src_nodes, Bt1, 40000);
    k_gather_dst<<<(4096 * 128 + 255) / 256, 256, 0, stream>>>(raw, d2d1, src_nodes, dst1, 4096);
    k_transpose_w<<<(256 * 128 + 255) / 256, 256, 0, stream>>>(w1, w1t);
    k_transpose_w<<<(256 * 128 + 255) / 256, 256, 0, stream>>>(w2, w2t);
    k_gemm_bf16<<<dim3(32, 10), 256, 0, stream>>>(dif1, Bt1, part1, 4096, 40000, 4000);
    k_reduce_cat<<<(4096 * 128 + 255) / 256, 256, 0, stream>>>(part1, 10, dst1, Acat1, 4096);
    k_mlp<<<32, 256, 0, stream>>>(Acat1, w1t, h1, 4096);

    // layer 2: gather from h1 + GEMM (M=1024, K=4096, split-K=4) + MLP
    k_gather_bt <<<(4096 + 255) / 256, 256, 0, stream>>>(h1, d2s2, nullptr, Bt2, 4096);
    k_gather_dst<<<(1024 * 128 + 255) / 256, 256, 0, stream>>>(h1, d2d2, nullptr, dst2, 1024);
    k_gemm_bf16<<<dim3(8, 4), 256, 0, stream>>>(dif2, Bt2, part2, 1024, 4096, 1024);
    k_reduce_cat<<<(1024 * 128 + 255) / 256, 256, 0, stream>>>(part2, 4, dst2, Acat2, 1024);
    k_mlp<<<8, 256, 0, stream>>>(Acat2, w2t, h2, 1024);

    // classifier + softmax
    k_cls<<<(1024 + 255) / 256, 256, 0, stream>>>(h2, wcls, (float*)d_out, 1024);
}
